// Attention_64115271794774
// MI455X (gfx1250) — matrix operand order repaired and checked by reading, not compiled, not run on hardware
//
#include <hip/hip_runtime.h>

// Causal GQA flash-attention for gfx1250 (wave32, WMMA f16->f32).
// Block = 128 threads = 4 waves = the 4 query heads sharing one KV head,
// all on the same 16-row q tile. K/V staged once in LDS (f16, converted
// once), double-buffered; V stored transposed so PV B-fragments are
// contiguous ds_load_b128 pairs. Causal mask is applied only in the single
// diagonal block; the steady-state loop is mask-free.

#define NUM_TOKENS   2048
#define NUM_HEADS    32
#define NUM_KV_HEADS 8
#define HEAD_DIM     128
#define Q_PER_KV     (NUM_HEADS / NUM_KV_HEADS)
#define Q_STRIDE     (NUM_HEADS * HEAD_DIM)
#define KV_STRIDE    (NUM_KV_HEADS * HEAD_DIM)

#define KSTR 136   // padded halves per K row   (272 B, 16B-aligned, banks spread)
#define VSTR 40    // padded halves per Vt row  (80 B, 16B-aligned, banks spread)

typedef __attribute__((ext_vector_type(16))) _Float16 v16h;
typedef __attribute__((ext_vector_type(8)))  _Float16 v8h;
typedef __attribute__((ext_vector_type(8)))  float    v8f;

__global__ __launch_bounds__(128) void fa_gqa_causal_kernel(
    const float* __restrict__ Q, const float* __restrict__ K,
    const float* __restrict__ V, float* __restrict__ O)
{
    __shared__ _Float16 ldsK[2][32 * KSTR];        // K block, row-major f16
    __shared__ _Float16 ldsV[2][HEAD_DIM * VSTR];  // V block, TRANSPOSED (d-major) f16

    const int tid  = threadIdx.x;
    const int lane = tid & 31;
    const int wave = tid >> 5;          // 0..3 -> which q head of this KV head
    const int m    = lane & 15;
    const int h    = lane >> 4;
    const int q0   = blockIdx.x * 16;
    const int kvh  = blockIdx.y;
    const int head = kvh * Q_PER_KV + wave;
    const float scale = 0.08838834764831845f;  // 128^-0.5, folded into Q

    const float* Kg = K + kvh * HEAD_DIM;
    const float* Vg = V + kvh * HEAD_DIM;

    // ---- Q as WMMA B-fragments (Q^T), per-wave head, loaded once ----
    // B layout (16-bit, 32x16): lane = column n, element e -> k = e + 8*h + (e>=8 ? 8 : 0)
    const float* qrow = Q + (q0 + m) * Q_STRIDE + head * HEAD_DIM;
    v16h qb[4];
#pragma unroll
    for (int c = 0; c < 4; ++c) {
#pragma unroll
        for (int e = 0; e < 16; ++e)
            qb[c][e] = (_Float16)(qrow[c * 32 + e + 8 * h + (e >= 8 ? 8 : 0)] * scale);
    }

    v8f oacc[8];
#pragma unroll
    for (int c = 0; c < 8; ++c) oacc[c] = (v8f){};

    float mrow = -1e30f, lrow = 0.0f;
    const int qglob  = q0 + m;
    const int kblast = ((q0 + 15) >> 5) << 5;  // the single block that needs masking

    // ---- cooperative stager: 32x128 f32 K and V -> f16 LDS (V transposed) ----
    auto stage = [&](int kb, int b) {
        {   // K: thread -> (row = tid/4, 32-wide d segment), coalesced b128 loads
            const int row = tid >> 2;
            const int dsg = (tid & 3) * 32;
            const float* src = Kg + (kb + row) * KV_STRIDE + dsg;
            if (kb + 63 < NUM_TOKENS)
                __builtin_prefetch(src + 32 * KV_STRIDE, 0, 3);  // global_prefetch_b8
            _Float16* dst = &ldsK[b][row * KSTR + dsg];
#pragma unroll
            for (int i = 0; i < 32; ++i) dst[i] = (_Float16)src[i];
        }
        {   // V: same coalesced loads, scatter-transpose to d-major
            const int row = tid & 31;
            const int dsg = (tid >> 5) * 32;
            const float* src = Vg + (kb + row) * KV_STRIDE + dsg;
            if (kb + 63 < NUM_TOKENS)
                __builtin_prefetch(src + 32 * KV_STRIDE, 0, 3);
#pragma unroll
            for (int i = 0; i < 32; ++i)
                ldsV[b][(dsg + i) * VSTR + row] = (_Float16)src[i];
        }
    };

    // ---- one 32-k block of flash attention; `masked` folds at compile time ----
    auto compute = [&](int kb, int b, bool masked) {
        // S^T tiles: st[t] = K[kb+16t..+15] x Q^T
        // A layout (16-bit, 16x32): lane = row m, element e -> k = e + 8h + (e>=8 ? 8 : 0)
        v8f st[2];
#pragma unroll
        for (int t = 0; t < 2; ++t) {
            v8f acc = (v8f){};
            const _Float16* krow = &ldsK[b][(t * 16 + m) * KSTR];
#pragma unroll
            for (int c = 0; c < 4; ++c) {
                const v8h lo = *(const v8h*)(krow + c * 32 + 8 * h);        // e = 0..7
                const v8h hi = *(const v8h*)(krow + c * 32 + 16 + 8 * h);   // e = 8..15
                v16h ka;
#pragma unroll
                for (int e = 0; e < 8; ++e) { ka[e] = lo[e]; ka[e + 8] = hi[e]; }
                acc = __builtin_amdgcn_wmma_f32_16x16x32_f16(
                    false, ka, false, qb[c], (short)0, acc, false, false);
            }
            st[t] = acc;
        }

        // causal mask only in the diagonal block
        if (masked) {
#pragma unroll
            for (int t = 0; t < 2; ++t) {
#pragma unroll
                for (int r = 0; r < 8; ++r) {
                    int kg = kb + t * 16 + 8 * h + r;
                    st[t][r] = (kg <= qglob) ? st[t][r] : -1e30f;
                }
            }
        }

        // per-row max (q row lives on the lane dim)
        float pm = -1e30f;
#pragma unroll
        for (int t = 0; t < 2; ++t)
#pragma unroll
            for (int r = 0; r < 8; ++r) pm = fmaxf(pm, st[t][r]);
        pm = fmaxf(pm, __shfl_xor(pm, 16, 32));
        float mnew  = fmaxf(mrow, pm);
        float alpha = __expf(mrow - mnew);

        // P = exp(S - m), already in PV A-fragment layout
        v16h pa;
        float ps = 0.0f;
#pragma unroll
        for (int r = 0; r < 8; ++r) {
            float p0 = __expf(st[0][r] - mnew);   // k = 8h + r
            float p1 = __expf(st[1][r] - mnew);   // k = 16 + 8h + r
            pa[r]     = (_Float16)p0;
            pa[r + 8] = (_Float16)p1;
            ps += p0 + p1;
        }
        ps  += __shfl_xor(ps, 16, 32);
        lrow = lrow * alpha + ps;
        mrow = mnew;

        // rescale O rows by alpha
#pragma unroll
        for (int r = 0; r < 8; ++r) {
            float ar = __shfl(alpha, r + 8 * h, 32);
#pragma unroll
            for (int c = 0; c < 8; ++c) oacc[c][r] *= ar;
        }

        // O += P * V : V^T in LDS makes B-fragments k-contiguous
        // B layout: lane = d column (c*16+m), element e -> k = e + 8h + (e>=8 ? 8 : 0)
#pragma unroll
        for (int c = 0; c < 8; ++c) {
            const _Float16* vcol = &ldsV[b][(c * 16 + m) * VSTR + 8 * h];
            const v8h lo = *(const v8h*)(vcol);      // k = 8h + 0..7
            const v8h hi = *(const v8h*)(vcol + 16); // k = 16 + 8h + 0..7
            v16h vbf;
#pragma unroll
            for (int e = 0; e < 8; ++e) { vbf[e] = lo[e]; vbf[e + 8] = hi[e]; }
            oacc[c] = __builtin_amdgcn_wmma_f32_16x16x32_f16(
                false, pa, false, vbf, (short)0, oacc[c], false, false);
        }
    };

    stage(0, 0);
    int buf = 0;

    // mask-free main loop: every block strictly below the diagonal
    for (int kb = 0; kb < kblast; kb += 32) {
        __syncthreads();                 // staged buf ready / old buf consumed
        stage(kb + 32, buf ^ 1);         // next block always exists here
        compute(kb, buf, false);
        buf ^= 1;
    }

    // single diagonal block with causal masking
    __syncthreads();
    compute(kblast, buf, true);

    // ---- finalize: divide by l per q-row and store ----
#pragma unroll
    for (int r = 0; r < 8; ++r) {
        float linv = 1.0f / __shfl(lrow, r + 8 * h, 32);
        float* orow = O + (q0 + r + 8 * h) * Q_STRIDE + head * HEAD_DIM + m;
#pragma unroll
        for (int c = 0; c < 8; ++c)
            orow[c * 16] = oacc[c][r] * linv;
    }
}

extern "C" void kernel_launch(void* const* d_in, const int* in_sizes, int n_in,
                              void* d_out, int out_size, void* d_ws, size_t ws_size,
                              hipStream_t stream) {
    const float* q = (const float*)d_in[0];
    const float* k = (const float*)d_in[1];
    const float* v = (const float*)d_in[2];
    float* out = (float*)d_out;
    (void)in_sizes; (void)n_in; (void)out_size; (void)d_ws; (void)ws_size;

    dim3 grid(NUM_TOKENS / 16, NUM_KV_HEADS);  // 128 q-tiles x 8 KV heads, 4 waves each
    fa_gqa_causal_kernel<<<grid, 128, 0, stream>>>(q, k, v, out);
}
